// HarmonyGenerator_17300128268571
// MI455X (gfx1250) — matrix operand order repaired and checked by reading, not compiled, not run on hardware
//
#include <hip/hip_runtime.h>

// ---------------------------------------------------------------------------
// HarmonyGenerator on MI455X (gfx1250): conv folded into weights, single
// f16 WMMA GEMM  [2048 x 50937] @ [50937 x 168], f32 accumulate.
// Round 3: 2 M-tiles per wave (M=32) -> halves L2 B-traffic and gives every
// B fragment two consuming WMMAs; region-uniform branch-free K loops.
// ---------------------------------------------------------------------------

typedef __attribute__((ext_vector_type(16))) _Float16 v16h;
typedef __attribute__((ext_vector_type(8)))  float    v8f;

#define D_IN      50937
#define D_LYR     50425
#define T_STEPS   2048
#define N_OUT     168
#define N_PAD     176          // 11 tiles of 16
#define N_TILES   11
#define KT_TOTAL  1592         // ceil(50937/32), K padded to 50944
#define K_CHUNKS  8
#define KT_PER_CHUNK 199       // 8 * 199 == 1592
#define M_PAIRS   64           // 64 waves x 32 rows = 2048

// workspace layout (bytes)
#define WS_CTX    0                         // 256 floats
#define WS_BIAS   1024                      // 176 floats
#define WS_BPACK  4096                      // 1592*11*512 halves = 17,932,288 B
#define WS_PART   (4096 + 17932288)         // K_CHUNKS*2048*176 floats = 11,534,336 B

// W = [w_chord | w_beat | w_mel] along N, each row-major (D_IN x n_i)
__device__ __forceinline__ float getW(int k, int n,
                                      const float* __restrict__ wc,
                                      const float* __restrict__ wbt,
                                      const float* __restrict__ wm) {
  if (n < 24) return wc[(size_t)k * 24 + n];
  if (n < 40) return wbt[(size_t)k * 16 + (n - 24)];
  return wm[(size_t)k * 128 + (n - 40)];
}

// --- prep 1: context embedding sum ------------------------------------------
__global__ __launch_bounds__(256) void ctx_kernel(const int* __restrict__ genre,
                                                  const int* __restrict__ tempo,
                                                  const int* __restrict__ key_sig,
                                                  const float* __restrict__ emb,
                                                  float* __restrict__ ctx) {
  int e = threadIdx.x;  // 0..255
  int i0 = genre[0];
  int i1 = 10 + tempo[0];
  int i2 = 20 + key_sig[0];
  int i3 = 34;  // length token (last slot)
  ctx[e] = emb[(size_t)i0 * 256 + e] + emb[(size_t)i1 * 256 + e] +
           emb[(size_t)i2 * 256 + e] + emb[(size_t)i3 * 256 + e];
}

// --- prep 2: bias'[n] = b[n] + conv_b * sum_d W[d,n] -------------------------
__global__ __launch_bounds__(256) void bias_kernel(const float* __restrict__ wc,
                                                   const float* __restrict__ wbt,
                                                   const float* __restrict__ wm,
                                                   const float* __restrict__ bc,
                                                   const float* __restrict__ bbt,
                                                   const float* __restrict__ bm,
                                                   const float* __restrict__ conv_b,
                                                   float* __restrict__ biasp) {
  __shared__ float red[256];
  int n = blockIdx.x;  // 0..167
  float s = 0.f;
  for (int d = threadIdx.x; d < D_IN; d += 256) s += getW(d, n, wc, wbt, wm);
  red[threadIdx.x] = s;
  __syncthreads();
  for (int off = 128; off > 0; off >>= 1) {
    if (threadIdx.x < off) red[threadIdx.x] += red[threadIdx.x + off];
    __syncthreads();
  }
  if (threadIdx.x == 0) {
    float b = (n < 24) ? bc[n] : (n < 40) ? bbt[n - 24] : bm[n - 40];
    biasp[n] = b + conv_b[0] * red[0];
  }
}

// --- prep 3: pack conv-folded W' into WMMA B-fragment layout -----------------
// B tile 32x16 f16, wave32: lane l holds column n = l&15, K
// (l>>4)*8 + (h&7) + (h>>3)*16. Stored contiguously per (k_tile, n_tile, lane) so GEMM
// B loads are 32B/lane coalesced b128 pairs.
__global__ __launch_bounds__(512) void bpack_kernel(const float* __restrict__ wc,
                                                    const float* __restrict__ wbt,
                                                    const float* __restrict__ wm,
                                                    const float* __restrict__ conv_w,
                                                    _Float16* __restrict__ bpack) {
  int kt = blockIdx.x;        // 0..1591
  int nt = blockIdx.y;        // 0..10
  int tid = threadIdx.x;      // 0..511
  int l = tid >> 4;           // lane 0..31
  int h = tid & 15;           // half index within fragment
  int k = kt * 32 + (l >> 4) * 8 + (h & 7) + (h >> 3) * 16;
  int n = nt * 16 + (l & 15);
  float v = 0.f;
  if (n < N_OUT && k < D_IN) {
    // middle row of the 3x3 kernel is the only one that touches real data
    float t0 = conv_w[3], t1 = conv_w[4], t2 = conv_w[5];
    v = t1 * getW(k, n, wc, wbt, wm);
    if (k + 1 < D_IN) v += t0 * getW(k + 1, n, wc, wbt, wm);
    if (k >= 1)       v += t2 * getW(k - 1, n, wc, wbt, wm);
  }
  bpack[((size_t)(kt * N_TILES + nt) * 32 + l) * 16 + h] = (_Float16)v;
}

// --- GEMM inner step: 2 A fragments vs 11 shared B fragments -----------------
// A tile 16x32 f16, wave32: lane row = l&15; lanes 0-15 hold K {0..7,16..23},
// lanes 16-31 hold K {8..15,24..31}.  s0/s1 point at this lane's first K
// element (region-uniform source) for the two M-tiles.
__device__ __forceinline__ void mma_step2(const float* __restrict__ s0,
                                          const float* __restrict__ s1,
                                          const _Float16* __restrict__ bb,
                                          v8f (&acc0)[N_TILES],
                                          v8f (&acc1)[N_TILES]) {
  v16h a0, a1;
#pragma unroll
  for (int i = 0; i < 8; ++i) { a0[i] = (_Float16)s0[i]; a0[8 + i] = (_Float16)s0[16 + i]; }
#pragma unroll
  for (int i = 0; i < 8; ++i) { a1[i] = (_Float16)s1[i]; a1[8 + i] = (_Float16)s1[16 + i]; }
#pragma unroll
  for (int nt = 0; nt < N_TILES; ++nt) {
    v16h b = *(const v16h*)(bb + (size_t)nt * 512);
    acc0[nt] = __builtin_amdgcn_wmma_f32_16x16x32_f16(
        false, a0, false, b, (short)0, acc0[nt], false, false);
    acc1[nt] = __builtin_amdgcn_wmma_f32_16x16x32_f16(
        false, a1, false, b, (short)0, acc1[nt], false, false);
  }
}

__device__ __forceinline__ void mma_step2_tail(const float* __restrict__ lrow0,
                                               const float* __restrict__ lrow1,
                                               int d0,
                                               const _Float16* __restrict__ bb,
                                               v8f (&acc0)[N_TILES],
                                               v8f (&acc1)[N_TILES]) {
  v16h a0, a1;
#pragma unroll
  for (int i = 0; i < 8; ++i) {
    int d = d0 + i;
    bool ok = d < D_IN;
    a0[i] = (_Float16)(ok ? lrow0[d - 512] : 0.f);
    a1[i] = (_Float16)(ok ? lrow1[d - 512] : 0.f);
  }
#pragma unroll
  for (int i = 0; i < 8; ++i) {
    int d = d0 + 16 + i;
    bool ok = d < D_IN;
    a0[8 + i] = (_Float16)(ok ? lrow0[d - 512] : 0.f);
    a1[8 + i] = (_Float16)(ok ? lrow1[d - 512] : 0.f);
  }
#pragma unroll
  for (int nt = 0; nt < N_TILES; ++nt) {
    v16h b = *(const v16h*)(bb + (size_t)nt * 512);
    acc0[nt] = __builtin_amdgcn_wmma_f32_16x16x32_f16(
        false, a0, false, b, (short)0, acc0[nt], false, false);
    acc1[nt] = __builtin_amdgcn_wmma_f32_16x16x32_f16(
        false, a1, false, b, (short)0, acc1[nt], false, false);
  }
}

// --- main GEMM: one wave = 32 rows (2 M-tiles) x one K chunk, all 11 N tiles -
// Region boundaries (256, 512) are 32-aligned, so each K tile lives in exactly
// one source array: kt 0..7 ctx, kt 8..15 melody, kt >= 16 lyrics. Only K
// chunk 0 touches the head regions; the dominant loop is branch-free.
__global__ __launch_bounds__(32) void gemm_kernel(const float* __restrict__ ctxv,
                                                  const float* __restrict__ melody,
                                                  const float* __restrict__ lyrics,
                                                  const _Float16* __restrict__ bpack,
                                                  float* __restrict__ part) {
  const int lane = threadIdx.x;   // 0..31
  const int mp   = blockIdx.x;    // 0..63 (pair of M tiles)
  const int kc   = blockIdx.y;    // 0..7
  const int row  = lane & 15;
  const int half = lane >> 4;
  const int t0   = mp * 32 + row;
  const int t1   = t0 + 16;
  const float* mrow0 = melody + (size_t)t0 * 256;
  const float* mrow1 = melody + (size_t)t1 * 256;
  const float* lrow0 = lyrics + (size_t)t0 * (size_t)D_LYR;
  const float* lrow1 = lyrics + (size_t)t1 * (size_t)D_LYR;

  v8f acc0[N_TILES], acc1[N_TILES];
  const v8f zero = {0.f, 0.f, 0.f, 0.f, 0.f, 0.f, 0.f, 0.f};
#pragma unroll
  for (int i = 0; i < N_TILES; ++i) { acc0[i] = zero; acc1[i] = zero; }

  const int kt0 = kc * KT_PER_CHUNK;
  const int kt1 = kt0 + KT_PER_CHUNK;
  const int ktFast = (kt1 > KT_TOTAL - 1) ? (KT_TOTAL - 1) : kt1;
  const int hoff = half * 8;

  int kt = kt0;
  if (kc == 0) {
    // ctx tiles: K 0..255 (ctx is row-independent -> both A fragments equal)
    for (; kt < 8; ++kt) {
      const float* s = ctxv + kt * 32 + hoff;
      mma_step2(s, s, bpack + (size_t)(kt * N_TILES) * 512 + (size_t)lane * 16,
                acc0, acc1);
    }
    // melody tiles: K 256..511
    for (; kt < 16; ++kt) {
      mma_step2(mrow0 + (kt * 32 - 256) + hoff, mrow1 + (kt * 32 - 256) + hoff,
                bpack + (size_t)(kt * N_TILES) * 512 + (size_t)lane * 16,
                acc0, acc1);
    }
  }
  // lyrics tiles: K 512.. (branch-free steady state, HBM stream)
  for (; kt < ktFast; ++kt) {
    const float* s0 = lrow0 + (kt * 32 - 512) + hoff;
    const float* s1 = lrow1 + (kt * 32 - 512) + hoff;
    __builtin_prefetch(s0 + 128, 0, 0);  // ~4 iterations ahead
    __builtin_prefetch(s1 + 128, 0, 0);
    mma_step2(s0, s1, bpack + (size_t)(kt * N_TILES) * 512 + (size_t)lane * 16,
              acc0, acc1);
  }

  if (kt1 > KT_TOTAL - 1) {  // last chunk owns the ragged final K tile
    const int ktt = KT_TOTAL - 1;
    mma_step2_tail(lrow0, lrow1, ktt * 32 + hoff,
                   bpack + (size_t)(ktt * N_TILES) * 512 + (size_t)lane * 16,
                   acc0, acc1);
  }

  // C/D layout: element v -> row v + 8*(lane>>4), col lane&15
  const int rb0 = mp * 32 + 8 * half;
  float* pw0 = part + ((size_t)kc * T_STEPS + rb0) * N_PAD + (lane & 15);
  float* pw1 = pw0 + (size_t)16 * N_PAD;
#pragma unroll
  for (int nt = 0; nt < N_TILES; ++nt) {
#pragma unroll
    for (int v = 0; v < 8; ++v) {
      pw0[(size_t)v * N_PAD + nt * 16] = acc0[nt][v];
      pw1[(size_t)v * N_PAD + nt * 16] = acc1[nt][v];
    }
  }
}

// --- deterministic reduction of K-chunk partials + bias ----------------------
__global__ __launch_bounds__(256) void reduce_kernel(const float* __restrict__ part,
                                                     const float* __restrict__ biasp,
                                                     float* __restrict__ out) {
  int idx = blockIdx.x * 256 + threadIdx.x;
  if (idx >= T_STEPS * N_OUT) return;
  int t = idx / N_OUT;
  int n = idx - t * N_OUT;
  float s = biasp[n];
#pragma unroll
  for (int c = 0; c < K_CHUNKS; ++c)
    s += part[((size_t)c * T_STEPS + t) * N_PAD + n];
  out[idx] = s;
}

extern "C" void kernel_launch(void* const* d_in, const int* in_sizes, int n_in,
                              void* d_out, int out_size, void* d_ws, size_t ws_size,
                              hipStream_t stream) {
  const int*   genre   = (const int*)d_in[0];
  const int*   tempo   = (const int*)d_in[1];
  const int*   key_sig = (const int*)d_in[2];
  // d_in[3] = length (constant 16, unused)
  const float* melody  = (const float*)d_in[4];
  const float* lyrics  = (const float*)d_in[5];
  const float* emb     = (const float*)d_in[6];
  const float* conv_w  = (const float*)d_in[7];
  const float* conv_b  = (const float*)d_in[8];
  const float* w_chord = (const float*)d_in[9];
  const float* b_chord = (const float*)d_in[10];
  const float* w_beat  = (const float*)d_in[11];
  const float* b_beat  = (const float*)d_in[12];
  const float* w_mel   = (const float*)d_in[13];
  const float* b_mel   = (const float*)d_in[14];
  float* out = (float*)d_out;

  char* ws = (char*)d_ws;
  float*    ctx   = (float*)(ws + WS_CTX);
  float*    biasp = (float*)(ws + WS_BIAS);
  _Float16* bpack = (_Float16*)(ws + WS_BPACK);
  float*    part  = (float*)(ws + WS_PART);

  hipLaunchKernelGGL(ctx_kernel, dim3(1), dim3(256), 0, stream,
                     genre, tempo, key_sig, emb, ctx);
  hipLaunchKernelGGL(bias_kernel, dim3(N_OUT), dim3(256), 0, stream,
                     w_chord, w_beat, w_mel, b_chord, b_beat, b_mel, conv_b, biasp);
  hipLaunchKernelGGL(bpack_kernel, dim3(KT_TOTAL, N_TILES), dim3(512), 0, stream,
                     w_chord, w_beat, w_mel, conv_w, bpack);
  hipLaunchKernelGGL(gemm_kernel, dim3(M_PAIRS, K_CHUNKS), dim3(32), 0, stream,
                     ctx, melody, lyrics, bpack, part);
  hipLaunchKernelGGL(reduce_kernel, dim3((T_STEPS * N_OUT + 255) / 256), dim3(256),
                     0, stream, part, biasp, out);
}
